// BModule_38671885534054
// MI455X (gfx1250) — matrix operand order repaired and checked by reading, not compiled, not run on hardware
//
#include <hip/hip_runtime.h>
#include <hip/hip_bf16.h>

// ---------------------------------------------------------------------------
// Problem constants (from the reference)
// ---------------------------------------------------------------------------
constexpr int NB = 4;      // batch
constexpr int TT = 1024;   // tokens
constexpr int DD = 512;    // model dim
constexpr int RR = 64;     // routing rank
constexpr int S0 = 1024, S1 = 256, S2 = 64;
constexpr int TOPK = 16;

typedef __attribute__((ext_vector_type(16))) __bf16 v16bf;
typedef __attribute__((ext_vector_type(8)))  __bf16 v8bf;
typedef __attribute__((ext_vector_type(8)))  float  v8f;
typedef __attribute__((ext_vector_type(4)))  float  v4f;

__device__ __forceinline__ v8bf cvt8(v4f a, v4f b) {
  v8bf r;
#pragma unroll
  for (int i = 0; i < 4; ++i) { r[i] = (__bf16)a[i]; r[i + 4] = (__bf16)b[i]; }
  return r;
}

// ---------------------------------------------------------------------------
// bf16 WMMA GEMM:  C[b] (+)= alpha * sigmoid(*gate) * A[b] @ (BT ? B[b]^T : B[b])
//
// Block: 128 threads = 4 waves. Block tile 32(M) x 64(N), K-step 32.
// Wave (wm, wn2) owns a 16x32 output strip: one A fragment + two B fragments
// -> 2 WMMAs per K-step with A reuse.
//
// Tiles are staged in LDS **already in WMMA fragment layout**:
//   A (16x32 bf16):  lane = (m&15) + 16*(k>>4),     elem = k & 15
//   B (32x16 bf16):  lane = (n&15) + 16*(k>>4),     elem = k & 15
// so each lane's fragment is 32 contiguous LDS bytes (2x ds_load_b128), and
// staging stores are 16B ds_store_b128 of packed cvt_pk_bf16 results.
// All M % 32 == 0, N % 64 == 0, K % 32 == 0 for this problem.
// ---------------------------------------------------------------------------
template <bool BT, bool ACC>
__global__ __launch_bounds__(128)
void gemm_bf16_kernel(const float* __restrict__ A, long sA, int lda,
                      const float* __restrict__ Bm, long sB, int ldb,
                      float* __restrict__ C, long sC, int ldc,
                      int M, int N, int K, float alpha,
                      const float* __restrict__ gate) {
  __shared__ __bf16 Afrag[2][32][16];   // [m-subtile][lane][elem]
  __shared__ __bf16 Bfrag[4][32][16];   // [n-subtile][lane][elem]

  const int b = blockIdx.z;
  const float* Ab = A + (long)b * sA;
  const float* Bb = Bm + (long)b * sB;
  float* Cb = C + (long)b * sC;

  const int row0 = blockIdx.y * 32;
  const int col0 = blockIdx.x * 64;
  const int tid  = threadIdx.x;
  const int lane = tid & 31;
  const int wave = tid >> 5;
  const int wsubm = wave >> 1;          // A subtile (rows wsubm*16..+15)
  const int wsubn = (wave & 1) * 2;     // B subtiles wsubn, wsubn+1

  // A staging: thread -> (row ar, 8 consecutive k at ac0)
  const int ar    = tid >> 2;           // 0..31
  const int ac0   = (tid & 3) * 8;      // 0,8,16,24
  const int a_sub = ar >> 4;
  const int a_lane = (ar & 15) + ((ac0 >> 4) << 4);
  const int a_e0   = ac0 & 15;          // 0 or 8

  // B staging: thread -> (column bc, 16 consecutive k at bk0)
  const int bc    = tid & 63;           // 0..63
  const int bk0   = (tid >> 6) * 16;    // 0 or 16
  const int b_sub = bc >> 4;
  const int b_lane = (bc & 15) + ((bk0 >> 4) << 4);

  v8f acc0 = {}, acc1 = {};

  for (int k0 = 0; k0 < K; k0 += 32) {
    // ---- stage A tile (32 rows x 32 k): 2x b128 load, 1x b128 LDS store ----
    {
      const float* src = Ab + (long)(row0 + ar) * lda + (k0 + ac0);
      v4f f0 = *(const v4f*)(src);
      v4f f1 = *(const v4f*)(src + 4);
      *(v8bf*)&Afrag[a_sub][a_lane][a_e0] = cvt8(f0, f1);
    }
    // ---- stage B tile (32 k x 64 n) ----
    if (BT) {
      // B stored [N,K]: a fragment column is 16 contiguous floats.
      const float* src = Bb + (long)(col0 + bc) * ldb + (k0 + bk0);
      v4f f0 = *(const v4f*)(src);
      v4f f1 = *(const v4f*)(src + 4);
      v4f f2 = *(const v4f*)(src + 8);
      v4f f3 = *(const v4f*)(src + 12);
      *(v8bf*)&Bfrag[b_sub][b_lane][0] = cvt8(f0, f1);
      *(v8bf*)&Bfrag[b_sub][b_lane][8] = cvt8(f2, f3);
    } else {
      // B stored [K,N]: strided column gather, coalesced across lanes.
      float tmp[16];
#pragma unroll
      for (int j = 0; j < 16; ++j)
        tmp[j] = Bb[(long)(k0 + bk0 + j) * ldb + (col0 + bc)];
      v8bf lo, hi;
#pragma unroll
      for (int j = 0; j < 8; ++j) {
        lo[j] = (__bf16)tmp[j];
        hi[j] = (__bf16)tmp[j + 8];
      }
      *(v8bf*)&Bfrag[b_sub][b_lane][0] = lo;
      *(v8bf*)&Bfrag[b_sub][b_lane][8] = hi;
    }
    __syncthreads();

    // ---- fragment loads: 32 contiguous bytes per lane ----
    v16bf afr = *(const v16bf*)&Afrag[wsubm][lane][0];
    v16bf bf0 = *(const v16bf*)&Bfrag[wsubn][lane][0];
    v16bf bf1 = *(const v16bf*)&Bfrag[wsubn + 1][lane][0];

    acc0 = __builtin_amdgcn_wmma_f32_16x16x32_bf16(false, afr, false, bf0,
                                                   (short)0, acc0, false, false);
    acc1 = __builtin_amdgcn_wmma_f32_16x16x32_bf16(false, afr, false, bf1,
                                                   (short)0, acc1, false, false);
    __syncthreads();
  }

  float g = 1.0f;
  if (gate) g = 1.0f / (1.0f + __expf(-gate[0]));
  const float scale = alpha * g;

  const int half = lane >> 4;
  const int l16  = lane & 15;
  const int rbase = row0 + wsubm * 16 + half * 8;
  const int c0 = col0 + wsubn * 16 + l16;
#pragma unroll
  for (int i = 0; i < 8; ++i) {
    long o0 = (long)(rbase + i) * ldc + c0;
    long o1 = o0 + 16;
    float u0 = scale * acc0[i];
    float u1 = scale * acc1[i];
    if (ACC) { u0 += Cb[o0]; u1 += Cb[o1]; }
    Cb[o0] = u0;
    Cb[o1] = u1;
  }
}

// ---------------------------------------------------------------------------
// Top-16 per row of logits [B, Ns, Nd] by iterative argmax in LDS.
// ---------------------------------------------------------------------------
__global__ __launch_bounds__(256)
void topk_kernel(const float* __restrict__ logits, int Ns, int Nd,
                 float* __restrict__ tkv, int* __restrict__ tki) {
  __shared__ float sv[1024];
  __shared__ float red[256];
  __shared__ int   redi[256];
  const int srow = blockIdx.x;
  const int b    = blockIdx.y;
  const int tid  = threadIdx.x;
  const float* row = logits + ((long)b * Ns + srow) * Nd;

  for (int j = tid; j < Nd; j += 256) sv[j] = row[j];
  __syncthreads();

  for (int t = 0; t < TOPK; ++t) {
    float best = -3.4e38f; int bi = 0;
    for (int j = tid; j < Nd; j += 256)
      if (sv[j] > best) { best = sv[j]; bi = j; }
    red[tid] = best; redi[tid] = bi;
    __syncthreads();
    for (int s = 128; s > 0; s >>= 1) {
      if (tid < s && red[tid + s] > red[tid]) {
        red[tid] = red[tid + s]; redi[tid] = redi[tid + s];
      }
      __syncthreads();
    }
    if (tid == 0) {
      long o = ((long)b * Ns + srow) * TOPK + t;
      tkv[o] = red[0];
      tki[o] = redi[0];
      sv[redi[0]] = -3.4e38f;
    }
    __syncthreads();
  }
}

// ---------------------------------------------------------------------------
// Edge weights + scatter:
//   w = sign(v) * softmax(|v|) over the 16 selected edges
//   c = w * softplus(src_state[s]) * sigmoid(gate)
//   dstate[idx] += c ; dval[idx,:] += c * src_val[s,:]   (atomics)
// ---------------------------------------------------------------------------
__global__ __launch_bounds__(256)
void edge_scatter_kernel(const float* __restrict__ tkv, const int* __restrict__ tki,
                         const float* __restrict__ src_state,
                         const float* __restrict__ src_val,
                         const float* __restrict__ gate,
                         float* __restrict__ dstate, float* __restrict__ dval,
                         int Ns, int Nd) {
  __shared__ float c[TOPK];
  __shared__ int   ix[TOPK];
  const int srow = blockIdx.x;
  const int b    = blockIdx.y;
  const int tid  = threadIdx.x;
  const long base16 = ((long)b * Ns + srow) * TOPK;

  if (tid == 0) {
    float av[TOPK], sg[TOPK];
    float mx = 0.0f;
    for (int e = 0; e < TOPK; ++e) {
      float v = tkv[base16 + e];
      sg[e] = (v > 0.0f) ? 1.0f : ((v < 0.0f) ? -1.0f : 0.0f);
      av[e] = fabsf(v);
      mx = fmaxf(mx, av[e]);
    }
    float sum = 0.0f;
    for (int e = 0; e < TOPK; ++e) { av[e] = __expf(av[e] - mx); sum += av[e]; }
    float st = src_state[(long)b * Ns + srow];
    float sp = (st > 20.0f) ? st : log1pf(__expf(st));  // softplus
    float g = 1.0f;
    if (gate) g = 1.0f / (1.0f + __expf(-gate[0]));
    float scale = sp * g / sum;
    for (int e = 0; e < TOPK; ++e) c[e] = sg[e] * av[e] * scale;
  }
  if (tid < TOPK) ix[tid] = tki[base16 + tid];
  __syncthreads();

  if (tid < TOPK) atomicAdd(&dstate[(long)b * Nd + ix[tid]], c[tid]);

  const float* sv = src_val + ((long)b * Ns + srow) * DD;
  for (int e = 0; e < TOPK; ++e) {
    float ce = c[e];
    float* dv = dval + ((long)b * Nd + ix[e]) * DD;
    for (int d = tid; d < DD; d += 256) atomicAdd(&dv[d], ce * sv[d]);
  }
}

// ---------------------------------------------------------------------------
// state <- sign(x) * softmax(|x|) over the slot axis, x = state + dstate.
// ---------------------------------------------------------------------------
__global__ __launch_bounds__(256)
void sstate_kernel(float* __restrict__ state, const float* __restrict__ dstate,
                   int N) {
  __shared__ float red[256];
  const int b = blockIdx.x, tid = threadIdx.x;
  float xv[4]; int cnt = 0;
  float amax = 0.0f;
  for (int n = tid; n < N; n += 256) {
    float x = state[(long)b * N + n] + dstate[(long)b * N + n];
    xv[cnt++] = x;
    amax = fmaxf(amax, fabsf(x));
  }
  red[tid] = amax; __syncthreads();
  for (int s = 128; s > 0; s >>= 1) {
    if (tid < s) red[tid] = fmaxf(red[tid], red[tid + s]);
    __syncthreads();
  }
  float mx = red[0]; __syncthreads();
  float lsum = 0.0f;
  for (int i = 0; i < cnt; ++i) lsum += __expf(fabsf(xv[i]) - mx);
  red[tid] = lsum; __syncthreads();
  for (int s = 128; s > 0; s >>= 1) {
    if (tid < s) red[tid] += red[tid + s];
    __syncthreads();
  }
  float inv = 1.0f / red[0];
  int i = 0;
  for (int n = tid; n < N; n += 256, ++i) {
    float x = xv[i];
    float sgn = (x > 0.0f) ? 1.0f : ((x < 0.0f) ? -1.0f : 0.0f);
    state[(long)b * N + n] = sgn * __expf(fabsf(x) - mx) * inv;
  }
}

// ---------------------------------------------------------------------------
// val <- layernorm(val + dval) * gamma + beta, per (b, slot) row of D=512.
// ---------------------------------------------------------------------------
__global__ __launch_bounds__(256)
void ln_kernel(float* __restrict__ val, const float* __restrict__ dval,
               const float* __restrict__ gamma, const float* __restrict__ beta,
               int N) {
  __shared__ float red[256];
  const int n = blockIdx.x, b = blockIdx.y, tid = threadIdx.x;
  const long base = ((long)b * N + n) * DD;
  float x0 = val[base + tid]       + dval[base + tid];
  float x1 = val[base + tid + 256] + dval[base + tid + 256];
  red[tid] = x0 + x1; __syncthreads();
  for (int s = 128; s > 0; s >>= 1) {
    if (tid < s) red[tid] += red[tid + s];
    __syncthreads();
  }
  float m = red[0] * (1.0f / 512.0f); __syncthreads();
  float d0 = x0 - m, d1 = x1 - m;
  red[tid] = d0 * d0 + d1 * d1; __syncthreads();
  for (int s = 128; s > 0; s >>= 1) {
    if (tid < s) red[tid] += red[tid + s];
    __syncthreads();
  }
  float inv = rsqrtf(red[0] * (1.0f / 512.0f) + 1e-5f);
  val[base + tid]       = d0 * inv * gamma[tid]       + beta[tid];
  val[base + tid + 256] = d1 * inv * gamma[tid + 256] + beta[tid + 256];
}

// ---------------------------------------------------------------------------
// In-place row softmax for read attention logits [B, Nrows, Nd].
// ---------------------------------------------------------------------------
__global__ __launch_bounds__(256)
void softmax_kernel(float* __restrict__ logits, int Nrows, int Nd) {
  __shared__ float red[256];
  const int r = blockIdx.x, b = blockIdx.y, tid = threadIdx.x;
  float* row = logits + ((long)b * Nrows + r) * Nd;
  float xv[4]; int cnt = 0;
  float mx = -3.4e38f;
  for (int j = tid; j < Nd; j += 256) { float x = row[j]; xv[cnt++] = x; mx = fmaxf(mx, x); }
  red[tid] = mx; __syncthreads();
  for (int s = 128; s > 0; s >>= 1) {
    if (tid < s) red[tid] = fmaxf(red[tid], red[tid + s]);
    __syncthreads();
  }
  mx = red[0]; __syncthreads();
  float lsum = 0.0f;
  for (int i = 0; i < cnt; ++i) { xv[i] = __expf(xv[i] - mx); lsum += xv[i]; }
  red[tid] = lsum; __syncthreads();
  for (int s = 128; s > 0; s >>= 1) {
    if (tid < s) red[tid] += red[tid + s];
    __syncthreads();
  }
  float inv = 1.0f / red[0];
  int i = 0;
  for (int j = tid; j < Nd; j += 256, ++i) row[j] = xv[i] * inv;
}

__global__ void zero_kernel(float* __restrict__ p, long n) {
  long i = (long)blockIdx.x * blockDim.x + threadIdx.x;
  long stride = (long)gridDim.x * blockDim.x;
  for (; i < n; i += stride) p[i] = 0.0f;
}

// ---------------------------------------------------------------------------
// Host-side orchestration helpers
// ---------------------------------------------------------------------------
static void gemm(const float* A, long sA, int lda,
                 const float* Bm, long sB, int ldb,
                 float* C, long sC, int ldc,
                 int M, int N, int K, float alpha,
                 const float* gate, bool bt, bool acc, hipStream_t st) {
  dim3 grid(N / 64, M / 32, NB), blk(128);
  if (bt) {
    if (acc) gemm_bf16_kernel<true, true><<<grid, blk, 0, st>>>(A, sA, lda, Bm, sB, ldb, C, sC, ldc, M, N, K, alpha, gate);
    else     gemm_bf16_kernel<true, false><<<grid, blk, 0, st>>>(A, sA, lda, Bm, sB, ldb, C, sC, ldc, M, N, K, alpha, gate);
  } else {
    if (acc) gemm_bf16_kernel<false, true><<<grid, blk, 0, st>>>(A, sA, lda, Bm, sB, ldb, C, sC, ldc, M, N, K, alpha, gate);
    else     gemm_bf16_kernel<false, false><<<grid, blk, 0, st>>>(A, sA, lda, Bm, sB, ldb, C, sC, ldc, M, N, K, alpha, gate);
  }
}

static void zero_buf(float* p, long n, hipStream_t st) {
  long g = (n + 255) / 256;
  if (g > 4096) g = 4096;
  zero_kernel<<<dim3((unsigned)g), 256, 0, st>>>(p, n);
}

static void do_transition(const float* src_val, const float* src_state, int Ns,
                          const float* dst_val, int Nd, const float* W,
                          const float* gate,
                          float* qb, float* kb, float* lg, float* tkv, int* tki,
                          float* dsb, float* dvb, hipStream_t st) {
  // q = src_val @ W[0] : [Ns, R]
  gemm(src_val, (long)Ns * DD, DD, W, 0, RR,
       qb, (long)Ns * RR, RR, Ns, RR, DD, 1.0f, nullptr, false, false, st);
  // kk = dst_val @ W[1] : [Nd, R]
  gemm(dst_val, (long)Nd * DD, DD, W + (long)DD * RR, 0, RR,
       kb, (long)Nd * RR, RR, Nd, RR, DD, 1.0f, nullptr, false, false, st);
  // logits = q @ kk^T / sqrt(R) : [Ns, Nd]
  gemm(qb, (long)Ns * RR, RR, kb, (long)Nd * RR, RR,
       lg, (long)Ns * Nd, Nd, Ns, Nd, RR, 0.125f, nullptr, true, false, st);
  topk_kernel<<<dim3(Ns, NB), 256, 0, st>>>(lg, Ns, Nd, tkv, tki);
  zero_buf(dsb, (long)NB * Nd, st);
  zero_buf(dvb, (long)NB * Nd * DD, st);
  edge_scatter_kernel<<<dim3(Ns, NB), 256, 0, st>>>(tkv, tki, src_state, src_val,
                                                    gate, dsb, dvb, Ns, Nd);
}

static void do_apply(float* state, float* val, const float* dsb, const float* dvb,
                     int N, const float* gamma, const float* beta, hipStream_t st) {
  sstate_kernel<<<NB, 256, 0, st>>>(state, dsb, N);
  ln_kernel<<<dim3(N, NB), 256, 0, st>>>(val, dvb, gamma, beta, N);
}

// ---------------------------------------------------------------------------
// Entry point
// ---------------------------------------------------------------------------
extern "C" void kernel_launch(void* const* d_in, const int* in_sizes, int n_in,
                              void* d_out, int out_size, void* d_ws, size_t ws_size,
                              hipStream_t stream) {
  (void)in_sizes; (void)n_in; (void)out_size; (void)ws_size;

  const float* tok_val    = (const float*)d_in[0];   // [B,T,D]
  const float* tok_state  = (const float*)d_in[1];   // [B,T]
  const float* mem_state0 = (const float*)d_in[2];   // [B,S0]
  const float* mem_val0   = (const float*)d_in[3];   // [B,S0,D]
  const float* mem_state1 = (const float*)d_in[4];
  const float* mem_val1   = (const float*)d_in[5];
  const float* mem_state2 = (const float*)d_in[6];
  const float* mem_val2   = (const float*)d_in[7];
  const float* write_route = (const float*)d_in[8];  // [1,2,D,R]
  const float* prop_route  = (const float*)d_in[9];  // [3,2,D,R]
  const float* level_route = (const float*)d_in[10]; // [2,2,D,R]
  const float* skip_route  = (const float*)d_in[11]; // [2,2,D,R]
  const float* skip_gates  = (const float*)d_in[12]; // [2]
  const float* ln_gamma    = (const float*)d_in[13]; // [3,D]
  const float* ln_beta     = (const float*)d_in[14]; // [3,D]
  const float* read_route  = (const float*)d_in[15]; // [3,2,D,R]
  const float* read_proj   = (const float*)d_in[16]; // [3,D,D]
  const float* read_gates  = (const float*)d_in[17]; // [3]
  float* out = (float*)d_out;                        // [B,T,D]

  // Workspace arena (f32, 256B aligned chunks).
  char* wptr = (char*)d_ws;
  auto alloc = [&](size_t nfl) {
    float* p = (float*)wptr;
    wptr += ((nfl * sizeof(float) + 255) / 256) * 256;
    return p;
  };
  float* v0  = alloc((size_t)NB * S0 * DD);
  float* s0  = alloc((size_t)NB * S0);
  float* v1  = alloc((size_t)NB * S1 * DD);
  float* s1  = alloc((size_t)NB * S1);
  float* v2  = alloc((size_t)NB * S2 * DD);
  float* s2  = alloc((size_t)NB * S2);
  float* qb  = alloc((size_t)NB * TT * RR);
  float* kb  = alloc((size_t)NB * TT * RR);
  float* lg  = alloc((size_t)NB * TT * S0);
  float* tkv = alloc((size_t)NB * TT * TOPK);
  int*   tki = (int*)alloc((size_t)NB * TT * TOPK);
  float* dsb = alloc((size_t)NB * S0);
  float* dvb = alloc((size_t)NB * S0 * DD);
  float* rb  = alloc((size_t)NB * TT * DD);

  // Fresh state copies (graph-capture-safe d2d copies).
  hipMemcpyAsync(s0, mem_state0, (size_t)NB * S0 * 4, hipMemcpyDeviceToDevice, stream);
  hipMemcpyAsync(v0, mem_val0, (size_t)NB * S0 * DD * 4, hipMemcpyDeviceToDevice, stream);
  hipMemcpyAsync(s1, mem_state1, (size_t)NB * S1 * 4, hipMemcpyDeviceToDevice, stream);
  hipMemcpyAsync(v1, mem_val1, (size_t)NB * S1 * DD * 4, hipMemcpyDeviceToDevice, stream);
  hipMemcpyAsync(s2, mem_state2, (size_t)NB * S2 * 4, hipMemcpyDeviceToDevice, stream);
  hipMemcpyAsync(v2, mem_val2, (size_t)NB * S2 * DD * 4, hipMemcpyDeviceToDevice, stream);
  hipMemcpyAsync(out, tok_val, (size_t)NB * TT * DD * 4, hipMemcpyDeviceToDevice, stream);

  const long WSTRIDE = (long)2 * DD * RR;  // one [2,D,R] routing block

  // ---- level 0: token write + propagation ----
  do_transition(tok_val, tok_state, TT, v0, S0, write_route, nullptr,
                qb, kb, lg, tkv, tki, dsb, dvb, stream);
  do_apply(s0, v0, dsb, dvb, S0, ln_gamma, ln_beta, stream);
  do_transition(v0, s0, S0, v0, S0, prop_route, nullptr,
                qb, kb, lg, tkv, tki, dsb, dvb, stream);
  do_apply(s0, v0, dsb, dvb, S0, ln_gamma, ln_beta, stream);

  // ---- level 1: up-transition, gated token skip, propagation ----
  do_transition(v0, s0, S0, v1, S1, level_route, nullptr,
                qb, kb, lg, tkv, tki, dsb, dvb, stream);
  do_apply(s1, v1, dsb, dvb, S1, ln_gamma + DD, ln_beta + DD, stream);
  do_transition(tok_val, tok_state, TT, v1, S1, skip_route, skip_gates + 0,
                qb, kb, lg, tkv, tki, dsb, dvb, stream);
  do_apply(s1, v1, dsb, dvb, S1, ln_gamma + DD, ln_beta + DD, stream);
  do_transition(v1, s1, S1, v1, S1, prop_route + WSTRIDE, nullptr,
                qb, kb, lg, tkv, tki, dsb, dvb, stream);
  do_apply(s1, v1, dsb, dvb, S1, ln_gamma + DD, ln_beta + DD, stream);

  // ---- level 2: up-transition, gated level-0 skip, propagation ----
  do_transition(v1, s1, S1, v2, S2, level_route + WSTRIDE, nullptr,
                qb, kb, lg, tkv, tki, dsb, dvb, stream);
  do_apply(s2, v2, dsb, dvb, S2, ln_gamma + 2 * DD, ln_beta + 2 * DD, stream);
  do_transition(v0, s0, S0, v2, S2, skip_route + WSTRIDE, skip_gates + 1,
                qb, kb, lg, tkv, tki, dsb, dvb, stream);
  do_apply(s2, v2, dsb, dvb, S2, ln_gamma + 2 * DD, ln_beta + 2 * DD, stream);
  do_transition(v2, s2, S2, v2, S2, prop_route + 2 * WSTRIDE, nullptr,
                qb, kb, lg, tkv, tki, dsb, dvb, stream);
  do_apply(s2, v2, dsb, dvb, S2, ln_gamma + 2 * DD, ln_beta + 2 * DD, stream);

  // ---- read: tokens attend over each memory level ----
  const float* mvs[3] = { v0, v1, v2 };
  const int ns[3] = { S0, S1, S2 };
  for (int l = 0; l < 3; ++l) {
    const int N = ns[l];
    const float* rw = read_route + (size_t)l * WSTRIDE;
    // q = tok_val @ read_route[l,0]
    gemm(tok_val, (long)TT * DD, DD, rw, 0, RR,
         qb, (long)TT * RR, RR, TT, RR, DD, 1.0f, nullptr, false, false, stream);
    // kk = mv @ read_route[l,1]
    gemm(mvs[l], (long)N * DD, DD, rw + (long)DD * RR, 0, RR,
         kb, (long)N * RR, RR, N, RR, DD, 1.0f, nullptr, false, false, stream);
    // logits = q @ kk^T / sqrt(R)
    gemm(qb, (long)TT * RR, RR, kb, (long)N * RR, RR,
         lg, (long)TT * N, N, TT, N, RR, 0.125f, nullptr, true, false, stream);
    softmax_kernel<<<dim3(TT, NB), 256, 0, stream>>>(lg, TT, N);
    // r = attn @ mv
    gemm(lg, (long)TT * N, N, mvs[l], (long)N * DD, DD,
         rb, (long)TT * DD, DD, TT, DD, N, 1.0f, nullptr, false, false, stream);
    // out += sigmoid(read_gates[l]) * (r @ read_proj[l])
    gemm(rb, (long)TT * DD, DD, read_proj + (size_t)l * DD * DD, 0, DD,
         out, (long)TT * DD, DD, TT, DD, DD, 1.0f, read_gates + l, false, true, stream);
  }
}
